// FeatureAlign_30056181138016
// MI455X (gfx1250) — matrix operand order repaired and checked by reading, not compiled, not run on hardware
//
#include <hip/hip_runtime.h>
#include <hip/hip_bf16.h>
#include <math.h>

typedef __attribute__((ext_vector_type(16))) _Float16 v16h;
typedef __attribute__((ext_vector_type(2)))  _Float16 v2h;
typedef __attribute__((ext_vector_type(8)))  float    v8f;

#define BDIM 256
#define NPIX (8 * 64 * 64)

// ---------------------------------------------------------------------------
// Fragment layouts (wave32, v_wmma_f32_16x16x32_f16), as packed by this code:
//  A (16x32 MxK): lane l (l<16): M=l, K = kb+j (j 0..15)
//                 lanes 16-31: same M, kb += 16.
//  B (32x16 KxN): lane l: N=l&15, K = k0+j (lanes 0-15) / k0+16+j (16-31).
//  C/D (16x16):   lane l: N=l&15, VGPR i -> M = i + (l>=16 ? 8 : 0).
// We pre-pack A (global, f16) and B (LDS, f16) so each fragment is one
// aligned 32-byte per-lane load: [tile][lane 0..31][16 halfs].
// ---------------------------------------------------------------------------

__device__ __forceinline__ v16h load_a_pk(const _Float16* __restrict__ Ap,
                                          int ktiles, int mt, int kt) {
  const int l = threadIdx.x & 31;
  return *(const v16h*)(Ap + ((mt * ktiles + kt) * 32 + l) * 16);
}

__device__ __forceinline__ v16h load_b_pk(const _Float16* __restrict__ Bs, int kt) {
  const int l = threadIdx.x & 31;
  return *(const v16h*)(Bs + kt * 512 + l * 16);
}

// Store B-tile element pair (rows k, k+1; k even) for pixel column n.
// Rows k,k+1 map to the same lane and adjacent halfs -> one 4-byte store.
__device__ __forceinline__ void store_b_pair(_Float16* __restrict__ Bs,
                                             int k, int n, float v0, float v1) {
  const int kt = k >> 5, kk = k & 31;
  const int lane = n + ((kk >> 4) << 4);
  _Float16* p = Bs + kt * 512 + lane * 16 + (kk & 15);
  *(v2h*)p = (v2h){(_Float16)v0, (_Float16)v1};
}

__device__ __forceinline__ void store_b_one(_Float16* __restrict__ Bs,
                                            int k, int n, float v) {
  const int kt = k >> 5, kk = k & 31;
  const int lane = n + ((kk >> 4) << 4);
  Bs[kt * 512 + lane * 16 + (kk & 15)] = (_Float16)v;
}

__device__ __forceinline__ int iclampi(int v, int lo, int hi) {
  return v < lo ? lo : (v > hi ? hi : v);
}

// ---- Weight pack: row-major f32 [M x K] -> f16 A-fragment layout ----------
// dst[((mt*ktiles + kt)*32 + lane)*16 + j]; rows >= M zero-padded.
__global__ void __launch_bounds__(BDIM) k_pack(const float* __restrict__ A,
                                               _Float16* __restrict__ dst,
                                               int M, int K, int mtiles) {
  const int ktiles = K >> 5;
  const int total = mtiles * ktiles * 512;
  for (int idx = blockIdx.x * BDIM + threadIdx.x; idx < total;
       idx += gridDim.x * BDIM) {
    const int frag = idx >> 9, within = idx & 511;
    const int lane = within >> 4, j = within & 15;
    const int mt = frag / ktiles, kt = frag - mt * ktiles;
    const int m = mt * 16 + (lane & 15);
    const int kb = kt * 32 + ((lane >> 4) << 4);
    const int k = kb + j;
    dst[idx] = (m < M) ? (_Float16)A[m * K + k] : (_Float16)0.0f;
  }
}

// ---- Kernel 1: feat_arm = relu(bn(W_fsm @ feat_l))  (1x1 conv, K=128) -----
__global__ void __launch_bounds__(BDIM) k_arm(const float* __restrict__ feat_l,
                                              const _Float16* __restrict__ Wp,
                                              const float* __restrict__ s_fsm,
                                              const float* __restrict__ b_fsm,
                                              float* __restrict__ feat_arm) {
  __shared__ __align__(32) _Float16 Bs[4 * 512];   // K=128 -> 4 ktiles
  const int tid = threadIdx.x;
  const int p0  = blockIdx.x * 16;
  for (int e = tid; e < 64 * 16; e += BDIM) {      // 64 row-pairs x 16 cols
    const int k = (e >> 4) << 1, n = e & 15;
    const int p = p0 + n, b = p >> 12, hw = p & 4095;
    const float v0 = feat_l[(b * 128 + k) * 4096 + hw];
    const float v1 = feat_l[(b * 128 + k + 1) * 4096 + hw];
    store_b_pair(Bs, k, n, v0, v1);
  }
  __syncthreads();
  const int wave = tid >> 5;
  v8f acc = {};
#pragma unroll
  for (int kt = 0; kt < 4; ++kt) {
    v16h a = load_a_pk(Wp, 4, wave, kt);
    v16h b = load_b_pk(Bs, kt);
    acc = __builtin_amdgcn_wmma_f32_16x16x32_f16(false, a, false, b, (short)0, acc, false, false);
  }
  const int l = tid & 31, n = l & 15, mo = (l >> 4) << 3;
  const int p = p0 + n, b = p >> 12, hw = p & 4095;
#pragma unroll
  for (int i = 0; i < 8; ++i) {
    const int m = wave * 16 + mo + i;
    feat_arm[(b * 128 + m) * 4096 + hw] = fmaxf(acc[i] * s_fsm[m] + b_fsm[m], 0.0f);
  }
}

// ---- Kernel 2: offset = relu(bn(W_off @ [feat_arm ; 2*up(feat_s)])) K=256 -
__global__ void __launch_bounds__(BDIM) k_off(const float* __restrict__ feat_arm,
                                              const float* __restrict__ feat_s,
                                              const _Float16* __restrict__ Wp,
                                              const float* __restrict__ s_off,
                                              const float* __restrict__ b_off,
                                              float* __restrict__ offset) {
  __shared__ __align__(32) _Float16 Bs[8 * 512];   // K=256 -> 8 ktiles
  const int tid = threadIdx.x;
  const int p0  = blockIdx.x * 16;
  for (int e = tid; e < 128 * 16; e += BDIM) {
    const int k = (e >> 4) << 1, n = e & 15;
    const int p = p0 + n, b = p >> 12, hw = p & 4095;
    const int h = hw >> 6, w = hw & 63;
    float v0, v1;
    if (k < 128) {
      v0 = feat_arm[(b * 128 + k) * 4096 + hw];
      v1 = feat_arm[(b * 128 + k + 1) * 4096 + hw];
    } else {  // nearest x2 upsample of feat_s, scaled by 2
      const int sb = ((b * 128 + (k - 128)) * 32 + (h >> 1)) * 32 + (w >> 1);
      v0 = 2.0f * feat_s[sb];
      v1 = 2.0f * feat_s[sb + 1024];
    }
    store_b_pair(Bs, k, n, v0, v1);
  }
  __syncthreads();
  const int wave = tid >> 5;
  v8f acc = {};
#pragma unroll
  for (int kt = 0; kt < 8; ++kt) {
    v16h a = load_a_pk(Wp, 8, wave, kt);
    v16h b = load_b_pk(Bs, kt);
    acc = __builtin_amdgcn_wmma_f32_16x16x32_f16(false, a, false, b, (short)0, acc, false, false);
  }
  const int l = tid & 31, n = l & 15, mo = (l >> 4) << 3;
  const int p = p0 + n, b = p >> 12, hw = p & 4095;
#pragma unroll
  for (int i = 0; i < 8; ++i) {
    const int m = wave * 16 + mo + i;
    offset[(b * 128 + m) * 4096 + hw] = fmaxf(acc[i] * s_off[m] + b_off[m], 0.0f);
  }
}

// ---- Kernel 3: om = conv3x3(offset, W_om) + b_om   (M=216->224, K=1152) ---
__global__ void __launch_bounds__(BDIM) k_om(const float* __restrict__ offset,
                                             const _Float16* __restrict__ Wp,
                                             const float* __restrict__ b_om,
                                             float* __restrict__ om) {
  __shared__ __align__(32) _Float16 Bs[36 * 512];  // K=1152 -> 36 ktiles
  const int tid = threadIdx.x;
  const int p0  = blockIdx.x * 16;
  __builtin_prefetch(Wp, 0, 1);
  // im2col tile: K index = c*9 + (kh*3+kw), matching W_om [216][128][3][3]
  for (int e = tid; e < 576 * 16; e += BDIM) {
    const int k = (e >> 4) << 1, n = e & 15;
    const int p = p0 + n, b = p >> 12, hw = p & 4095;
    const int h = hw >> 6, w = hw & 63;
    float v[2];
#pragma unroll
    for (int q = 0; q < 2; ++q) {
      const int kq = k + q;
      const int c = kq / 9, tap = kq - c * 9;
      const int hh = h + tap / 3 - 1, ww = w + (tap % 3) - 1;
      v[q] = (hh >= 0 && hh < 64 && ww >= 0 && ww < 64)
                 ? offset[(b * 128 + c) * 4096 + hh * 64 + ww] : 0.0f;
    }
    store_b_pair(Bs, k, n, v[0], v[1]);
  }
  __syncthreads();
  const int wave = tid >> 5;
  const bool two = (wave < 6);                      // M tiles 0..13
  v8f acc0 = {}, acc1 = {};
  for (int kt = 0; kt < 36; ++kt) {
    v16h bfrag = load_b_pk(Bs, kt);
    v16h a0 = load_a_pk(Wp, 36, wave, kt);
    acc0 = __builtin_amdgcn_wmma_f32_16x16x32_f16(false, a0, false, bfrag, (short)0, acc0, false, false);
    if (two) {
      v16h a1 = load_a_pk(Wp, 36, wave + 8, kt);
      acc1 = __builtin_amdgcn_wmma_f32_16x16x32_f16(false, a1, false, bfrag, (short)0, acc1, false, false);
    }
  }
  const int l = tid & 31, n = l & 15, mo = (l >> 4) << 3;
  const int p = p0 + n, b = p >> 12, hw = p & 4095;
#pragma unroll
  for (int i = 0; i < 8; ++i) {
    const int m = wave * 16 + mo + i;
    om[(b * 216 + m) * 4096 + hw] = acc0[i] + b_om[m];
  }
  if (two) {
#pragma unroll
    for (int i = 0; i < 8; ++i) {
      const int m = (wave + 8) * 16 + mo + i;
      if (m < 216) om[(b * 216 + m) * 4096 + hw] = acc1[i] + b_om[m];
    }
  }
}

// ---- Kernel 4: DCNv2 bilinear gather + grouped contraction (M=64, K=1152) -
__global__ void __launch_bounds__(BDIM) k_dcn(const float* __restrict__ feat_s,
                                              const float* __restrict__ om,
                                              const _Float16* __restrict__ Wp,
                                              const float* __restrict__ b_dcn,
                                              float* __restrict__ feat_align) {
  __shared__ __align__(32) _Float16 Bs[36 * 512];
  const int tid = threadIdx.x;
  const int p0  = blockIdx.x * 16;
  // 1152 (pixel, group, tap) tasks; each fills 16 channels of the val tile.
  for (int t = tid; t < 16 * 72; t += BDIM) {
    const int n = t / 72, gk = t - n * 72;
    const int g = gk / 9, k = gk - g * 9;
    const int p = p0 + n, b = p >> 12, hw = p & 4095;
    const int h = hw >> 6, w = hw & 63;
    const int obase = (b * 216 + gk) * 4096 + hw;
    const float oy = om[obase];
    const float ox = om[obase + 72 * 4096];
    float mm = om[obase + 144 * 4096];
    mm = 1.0f / (1.0f + __expf(-mm));
    const float py = (float)h + (float)(k / 3 - 1) + oy;
    const float px = (float)w + (float)(k % 3 - 1) + ox;
    const float fy0 = floorf(py), fx0 = floorf(px);
    const float ty = py - fy0, tx = px - fx0;
    const int iy0 = (int)fy0, ix0 = (int)fx0;
    const int iy1 = iy0 + 1,  ix1 = ix0 + 1;
    const bool vy0 = (iy0 >= 0) && (iy0 < 64), vy1 = (iy1 >= 0) && (iy1 < 64);
    const bool vx0 = (ix0 >= 0) && (ix0 < 64), vx1 = (ix1 >= 0) && (ix1 < 64);
    const int cy0 = iclampi(iy0, 0, 63) >> 1, cy1 = iclampi(iy1, 0, 63) >> 1;
    const int cx0 = iclampi(ix0, 0, 63) >> 1, cx1 = iclampi(ix1, 0, 63) >> 1;
    // feat_up[b,c,y,x] == feat_s[b,c,y>>1,x>>1] (nearest x2 upsample)
    const int o00 = cy0 * 32 + cx0, o01 = cy0 * 32 + cx1;
    const int o10 = cy1 * 32 + cx0, o11 = cy1 * 32 + cx1;
    const float w00 = (1.0f - ty) * (1.0f - tx) * ((vy0 && vx0) ? mm : 0.0f);
    const float w01 = (1.0f - ty) * tx          * ((vy0 && vx1) ? mm : 0.0f);
    const float w10 = ty * (1.0f - tx)          * ((vy1 && vx0) ? mm : 0.0f);
    const float w11 = ty * tx                   * ((vy1 && vx1) ? mm : 0.0f);
    const int cb = (b * 128 + g * 16) * 1024;
#pragma unroll
    for (int cg = 0; cg < 16; ++cg) {
      const float* fp = feat_s + cb + cg * 1024;
      const float v = w00 * fp[o00] + w01 * fp[o01] + w10 * fp[o10] + w11 * fp[o11];
      store_b_one(Bs, (g * 16 + cg) * 9 + k, n, v);   // K = c*9 + tap
    }
  }
  __syncthreads();
  const int wave = tid >> 5;
  if (wave < 4) {   // M=64 -> 4 tiles; wave-uniform branch keeps EXEC all-ones
    v8f acc = {};
    for (int kt = 0; kt < 36; ++kt) {
      v16h a = load_a_pk(Wp, 36, wave, kt);
      v16h b = load_b_pk(Bs, kt);
      acc = __builtin_amdgcn_wmma_f32_16x16x32_f16(false, a, false, b, (short)0, acc, false, false);
    }
    const int l = tid & 31, n = l & 15, mo = (l >> 4) << 3;
    const int p = p0 + n, b = p >> 12, hw = p & 4095;
#pragma unroll
    for (int i = 0; i < 8; ++i) {
      const int m = wave * 16 + mo + i;
      feat_align[(b * 64 + m) * 4096 + hw] = fmaxf(acc[i] + b_dcn[m], 0.0f);
    }
  }
}

// ---- Kernel 5: feat = relu(bn(W_cat @ feat_align)) + feat_arm  (K=64) -----
__global__ void __launch_bounds__(BDIM) k_cat(const float* __restrict__ feat_align,
                                              const float* __restrict__ feat_arm,
                                              const _Float16* __restrict__ Wp,
                                              const float* __restrict__ s_cat,
                                              const float* __restrict__ b_cat,
                                              float* __restrict__ feat_out) {
  __shared__ __align__(32) _Float16 Bs[2 * 512];   // K=64 -> 2 ktiles
  const int tid = threadIdx.x;
  const int p0  = blockIdx.x * 16;
  for (int e = tid; e < 32 * 16; e += BDIM) {
    const int k = (e >> 4) << 1, n = e & 15;
    const int p = p0 + n, b = p >> 12, hw = p & 4095;
    const float v0 = feat_align[(b * 64 + k) * 4096 + hw];
    const float v1 = feat_align[(b * 64 + k + 1) * 4096 + hw];
    store_b_pair(Bs, k, n, v0, v1);
  }
  __syncthreads();
  const int wave = tid >> 5;
  v8f acc = {};
#pragma unroll
  for (int kt = 0; kt < 2; ++kt) {
    v16h a = load_a_pk(Wp, 2, wave, kt);
    v16h b = load_b_pk(Bs, kt);
    acc = __builtin_amdgcn_wmma_f32_16x16x32_f16(false, a, false, b, (short)0, acc, false, false);
  }
  const int l = tid & 31, n = l & 15, mo = (l >> 4) << 3;
  const int p = p0 + n, b = p >> 12, hw = p & 4095;
#pragma unroll
  for (int i = 0; i < 8; ++i) {
    const int m = wave * 16 + mo + i;
    const int idx = (b * 128 + m) * 4096 + hw;
    const float v = fmaxf(acc[i] * s_cat[m] + b_cat[m], 0.0f);
    feat_out[idx] = v + feat_arm[idx];
  }
}

extern "C" void kernel_launch(void* const* d_in, const int* in_sizes, int n_in,
                              void* d_out, int out_size, void* d_ws, size_t ws_size,
                              hipStream_t stream) {
  const float* feat_l = (const float*)d_in[0];
  const float* feat_s = (const float*)d_in[1];
  const float* W_fsm  = (const float*)d_in[2];
  const float* s_fsm  = (const float*)d_in[3];
  const float* b_fsm  = (const float*)d_in[4];
  const float* W_off  = (const float*)d_in[5];
  const float* s_off  = (const float*)d_in[6];
  const float* b_off  = (const float*)d_in[7];
  const float* W_om   = (const float*)d_in[8];
  const float* b_om   = (const float*)d_in[9];
  const float* W_dcn  = (const float*)d_in[10];
  const float* b_dcn  = (const float*)d_in[11];
  const float* W_cat  = (const float*)d_in[12];
  const float* s_cat  = (const float*)d_in[13];
  const float* b_cat  = (const float*)d_in[14];

  float* out      = (float*)d_out;
  float* feat_out = out;                    // output 0: feat      [8,128,64,64]
  float* arm_out  = out + 8 * 128 * 4096;   // output 1: feat_arm  [8,128,64,64]

  float* ws        = (float*)d_ws;
  float* offset_ws = ws;                              // [8,128,64,64]
  float* om_ws     = offset_ws + 8 * 128 * 4096;      // [8,216,64,64]
  float* align_ws  = om_ws + 8 * 216 * 4096;          // [8, 64,64,64]

  // Packed f16 weights (A-fragment layout), after the float scratch.
  _Float16* wp     = (_Float16*)(align_ws + 8 * 64 * 4096);
  _Float16* Wp_fsm = wp;                        //  8 mt x  4 kt x 512
  _Float16* Wp_off = Wp_fsm +  8 * 4 * 512;     //  8 mt x  8 kt x 512
  _Float16* Wp_om  = Wp_off +  8 * 8 * 512;     // 14 mt x 36 kt x 512
  _Float16* Wp_dcn = Wp_om  + 14 * 36 * 512;    //  4 mt x 36 kt x 512
  _Float16* Wp_cat = Wp_dcn +  4 * 36 * 512;    //  8 mt x  2 kt x 512

  const dim3 block(BDIM);
  k_pack<<<dim3(( 8 *  4 * 512 + BDIM - 1) / BDIM), block, 0, stream>>>(W_fsm, Wp_fsm, 128,  128,  8);
  k_pack<<<dim3(( 8 *  8 * 512 + BDIM - 1) / BDIM), block, 0, stream>>>(W_off, Wp_off, 128,  256,  8);
  k_pack<<<dim3((14 * 36 * 512 + BDIM - 1) / BDIM), block, 0, stream>>>(W_om,  Wp_om,  216, 1152, 14);
  k_pack<<<dim3(( 4 * 36 * 512 + BDIM - 1) / BDIM), block, 0, stream>>>(W_dcn, Wp_dcn,  64, 1152,  4);
  k_pack<<<dim3(( 8 *  2 * 512 + BDIM - 1) / BDIM), block, 0, stream>>>(W_cat, Wp_cat, 128,   64,  8);

  const dim3 grid(NPIX / 16);
  k_arm<<<grid, block, 0, stream>>>(feat_l, Wp_fsm, s_fsm, b_fsm, arm_out);
  k_off<<<grid, block, 0, stream>>>(arm_out, feat_s, Wp_off, s_off, b_off, offset_ws);
  k_om <<<grid, block, 0, stream>>>(offset_ws, Wp_om, b_om, om_ws);
  k_dcn<<<grid, block, 0, stream>>>(feat_s, om_ws, Wp_dcn, b_dcn, align_ws);
  k_cat<<<grid, block, 0, stream>>>(align_ws, arm_out, Wp_cat, s_cat, b_cat, feat_out);
}
